// CausalSelfAttention_10694468567504
// MI455X (gfx1250) — matrix operand order repaired and checked by reading, not compiled, not run on hardware
//
#include <hip/hip_runtime.h>
#include <hip/hip_bf16.h>
#include <stdint.h>

// Problem constants (reference: B=8, T=1024, C=1024, H=16, D=64)
#define NB 8
#define NT 1024
#define NC 1024
#define NH 16
#define ND 64

typedef __attribute__((ext_vector_type(16))) __bf16 v16bf;
typedef __attribute__((ext_vector_type(8)))  __bf16 v8bf;
typedef __attribute__((ext_vector_type(4)))  __bf16 v4bf;
typedef __attribute__((ext_vector_type(8)))  float  v8f;
typedef int v4i __attribute__((vector_size(4 * sizeof(int))));

#if __has_builtin(__builtin_amdgcn_global_load_async_to_lds_b128)
#define HAVE_ASYNC_LDS 1
typedef __attribute__((address_space(1))) v4i* gptr_v4i;
typedef __attribute__((address_space(3))) v4i* lptr_v4i;
#else
#define HAVE_ASYNC_LDS 0
#endif

__device__ __forceinline__ void wait_async0() {
#if __has_builtin(__builtin_amdgcn_s_wait_asynccnt)
  __builtin_amdgcn_s_wait_asynccnt(0);
#else
  asm volatile("s_wait_asynccnt 0" ::: "memory");
#endif
}

__device__ __forceinline__ v8f vzero8() {
  v8f z = {0.f, 0.f, 0.f, 0.f, 0.f, 0.f, 0.f, 0.f};
  return z;
}

// Native conversion: backend emits v_cvt_pk_bf16_f32 for packed pairs.
__device__ __forceinline__ __bf16 f2bf(float f) { return (__bf16)f; }

// Combine two 16-byte runs into a 16-element fragment register set.
__device__ __forceinline__ v16bf mk_frag(const __bf16* p0, const __bf16* p1) {
  v8bf lo = *(const v8bf*)p0;
  v8bf hi = *(const v8bf*)p1;
  return __builtin_shufflevector(lo, hi, 0,1,2,3,4,5,6,7,8,9,10,11,12,13,14,15);
}

__device__ __forceinline__ v8f wmma_bf16(v16bf a, v16bf b, v8f c) {
  return __builtin_amdgcn_wmma_f32_16x16x32_bf16(false, a, false, b, (short)0, c,
                                                 false, false);
}

// A-matrix 16x32 bf16 per-lane K run:   [16h, 16h+15]  (two 8-elem loads)
// B-matrix 32x16 bf16 per-lane K run:   [16h, 16h+15]  (two 8-elem loads)

// ---------------------------------------------------------------------------
// Kernel 1: qkv = x @ w_attn + b_attn ; scatter to Q(scaled)/K [B,H,T,D] bf16
// and V transposed [B,H,D,T] bf16.
// Block: 128(M tokens) x 128(N cols), 8 waves -> each 32x64 (2x4 WMMA tiles).
// ---------------------------------------------------------------------------
__global__ __launch_bounds__(256)
void qkv_gemm_kernel(const float* __restrict__ x, const float* __restrict__ w,
                     const float* __restrict__ bias,
                     __bf16* __restrict__ Qo, __bf16* __restrict__ Ko,
                     __bf16* __restrict__ Vt) {
  __shared__ __attribute__((aligned(16))) __bf16 As[128][40];
  __shared__ __attribute__((aligned(16))) __bf16 Bs[128][40];
  const int tid = threadIdx.x;
  const int lane = tid & 31, wv = tid >> 5;
  const int hlf = lane >> 4, l15 = lane & 15;
  const int wm = wv >> 1, wn = wv & 1;
  const int row0 = blockIdx.x * 128;  // token rows (b*T+t)
  const int n0   = blockIdx.y * 128;  // output cols in [0, 3C)

  v8f acc[2][4];
#pragma unroll
  for (int i = 0; i < 2; i++)
#pragma unroll
    for (int j = 0; j < 4; j++) acc[i][j] = vzero8();

  for (int k0 = 0; k0 < NC; k0 += 32) {
    // Batch ALL staging loads first (8 x global_load_b128 in flight), then
    // convert + store -- keeps memory-level parallelism instead of
    // load->wait0->cvt->store serialization.
    float4 fa[4], fb[4];
#pragma unroll
    for (int i = 0; i < 4; i++) {
      int idx = tid + i * 256;            // 0..1023 float4 chunks of A
      fa[i] = *(const float4*)&x[(size_t)(row0 + (idx >> 3)) * NC + k0 +
                                 (idx & 7) * 4];
    }
#pragma unroll
    for (int i = 0; i < 4; i++) {
      int idx = tid + i * 256;            // 0..1023 float4 chunks of B
      fb[i] = *(const float4*)&w[(size_t)(k0 + (idx >> 5)) * (3 * NC) + n0 +
                                 (idx & 31) * 4];
    }
#pragma unroll
    for (int i = 0; i < 4; i++) {
      int idx = tid + i * 256;
      int r = idx >> 3, c = (idx & 7) * 4;
      v4bf o = { f2bf(fa[i].x), f2bf(fa[i].y), f2bf(fa[i].z), f2bf(fa[i].w) };
      *(v4bf*)&As[r][c] = o;
    }
#pragma unroll
    for (int i = 0; i < 4; i++) {
      int idx = tid + i * 256;
      int kk = idx >> 5, n4 = (idx & 31) * 4;
      Bs[n4 + 0][kk] = f2bf(fb[i].x);
      Bs[n4 + 1][kk] = f2bf(fb[i].y);
      Bs[n4 + 2][kk] = f2bf(fb[i].z);
      Bs[n4 + 3][kk] = f2bf(fb[i].w);
    }
    __syncthreads();
    if (k0 + 32 < NC) {
      __builtin_prefetch(&x[(size_t)(row0 + (tid >> 1)) * NC + k0 + 32], 0, 1);
      __builtin_prefetch(&w[(size_t)(k0 + 32 + lane) * (3 * NC) + n0], 0, 1);
    }
    v16bf af[2], bfr[4];
#pragma unroll
    for (int mt = 0; mt < 2; mt++) {
      int r = wm * 32 + mt * 16 + l15;
      af[mt] = mk_frag(&As[r][16 * hlf], &As[r][16 * hlf + 8]);
    }
#pragma unroll
    for (int nt = 0; nt < 4; nt++) {
      int n = wn * 64 + nt * 16 + l15;
      bfr[nt] = mk_frag(&Bs[n][16 * hlf], &Bs[n][16 * hlf + 8]);
    }
#pragma unroll
    for (int mt = 0; mt < 2; mt++)
#pragma unroll
      for (int nt = 0; nt < 4; nt++)
        acc[mt][nt] = wmma_bf16(af[mt], bfr[nt], acc[mt][nt]);
    __syncthreads();
  }

  // Epilogue: bias + scatter (fold softmax scale 1/sqrt(64) into Q)
#pragma unroll
  for (int mt = 0; mt < 2; mt++) {
#pragma unroll
    for (int nt = 0; nt < 4; nt++) {
      int ng = n0 + wn * 64 + nt * 16 + l15;
      int sec = ng >> 10;  // 0=Q 1=K 2=V
      int c1 = ng & 1023;
      int h = c1 >> 6, d = c1 & 63;
      float bv = bias[ng];
#pragma unroll
      for (int r = 0; r < 8; r++) {
        int mg = row0 + wm * 32 + mt * 16 + 8 * hlf + r;  // b*T + t
        int b = mg >> 10, t = mg & 1023;
        float val = acc[mt][nt][r] + bv;
        size_t hd = (size_t)(b * NH + h) * NT + t;
        if (sec == 0)      Qo[hd * ND + d] = f2bf(val * 0.125f);
        else if (sec == 1) Ko[hd * ND + d] = f2bf(val);
        else               Vt[((size_t)(b * NH + h) * ND + d) * NT + t] = f2bf(val);
      }
    }
  }
}

// ---------------------------------------------------------------------------
// Kernel 2: causal flash attention per (b,h). Block = 128 query rows, 8 waves,
// each wave owns 16 query rows and streams 32-wide key tiles.
// ---------------------------------------------------------------------------
__global__ __launch_bounds__(256)
void attention_kernel(const __bf16* __restrict__ Q, const __bf16* __restrict__ K,
                      const __bf16* __restrict__ Vt, __bf16* __restrict__ Y) {
  __shared__ __attribute__((aligned(16))) __bf16 Ps[8][16][40];
  const int tid = threadIdx.x, lane = tid & 31, wv = tid >> 5;
  const int hlf = lane >> 4, l15 = lane & 15;
  const int bh = blockIdx.y;
  const int b = bh >> 4, h = bh & 15;
  const int q_base = blockIdx.x * 128 + wv * 16;

  const __bf16* Qh = Q + (size_t)bh * NT * ND;
  const __bf16* Kh = K + (size_t)bh * NT * ND;
  const __bf16* Vh = Vt + (size_t)bh * ND * NT;

  // Q fragments (A-layout), kept in registers; 2 k-chunks cover D=64
  v16bf aq[2];
  {
    const __bf16* qr = Qh + (size_t)(q_base + l15) * ND;
#pragma unroll
    for (int kc = 0; kc < 2; kc++)
      aq[kc] = mk_frag(qr + kc * 32 + 16 * hlf, qr + kc * 32 + 16 * hlf + 8);
  }

  v8f O[4];
#pragma unroll
  for (int i = 0; i < 4; i++) O[i] = vzero8();
  float m_i[8], l_i[8];
#pragma unroll
  for (int r = 0; r < 8; r++) { m_i[r] = -1e30f; l_i[r] = 0.f; }

  const int kmax = q_base + 15;
  for (int k0 = 0; k0 <= kmax; k0 += 32) {
    // S = Q K^T (scale already folded into Q); two 16-col tiles
    v8f s[2];
#pragma unroll
    for (int nt = 0; nt < 2; nt++) {
      const __bf16* kr = Kh + (size_t)(k0 + nt * 16 + l15) * ND;
      v8f sa = vzero8();
#pragma unroll
      for (int kc = 0; kc < 2; kc++) {
        v16bf bk = mk_frag(kr + kc * 32 + 16 * hlf, kr + kc * 32 + 16 * hlf + 8);
        sa = wmma_bf16(aq[kc], bk, sa);
      }
      s[nt] = sa;
    }
    // Online softmax; row = q_base + 8*hlf + r, col = k0 + nt*16 + l15
    float alpha[8];
#pragma unroll
    for (int r = 0; r < 8; r++) {
      int qrow = q_base + 8 * hlf + r;
      int c0 = k0 + l15, c1 = c0 + 16;
      float v0 = (c0 <= qrow) ? s[0][r] : -1e30f;
      float v1 = (c1 <= qrow) ? s[1][r] : -1e30f;
      float mx = fmaxf(v0, v1);
#pragma unroll
      for (int msk = 8; msk >= 1; msk >>= 1) mx = fmaxf(mx, __shfl_xor(mx, msk, 32));
      float mnew = fmaxf(m_i[r], mx);
      float a = __expf(m_i[r] - mnew);
      float p0 = __expf(v0 - mnew);
      float p1 = __expf(v1 - mnew);
      float ps = p0 + p1;
#pragma unroll
      for (int msk = 8; msk >= 1; msk >>= 1) ps += __shfl_xor(ps, msk, 32);
      l_i[r] = l_i[r] * a + ps;
      m_i[r] = mnew;
      alpha[r] = a;
      Ps[wv][8 * hlf + r][l15]      = f2bf(p0);
      Ps[wv][8 * hlf + r][16 + l15] = f2bf(p1);
    }
#pragma unroll
    for (int dt = 0; dt < 4; dt++)
#pragma unroll
      for (int r = 0; r < 8; r++) O[dt][r] *= alpha[r];

    // Intra-wave LDS RAW: ensure P stores landed before fragment reload
    asm volatile("s_wait_dscnt 0" ::: "memory");
    v16bf pa;
    {
      const __bf16* pr = &Ps[wv][l15][0];
      pa = mk_frag(pr + 16 * hlf, pr + 16 * hlf + 8);
    }
    // O += P @ V ; V^T layout makes B-fragments contiguous along T
#pragma unroll
    for (int dt = 0; dt < 4; dt++) {
      const __bf16* vr = Vh + (size_t)(dt * 16 + l15) * NT + k0;
      v16bf bv = mk_frag(vr + 16 * hlf, vr + 16 * hlf + 8);
      O[dt] = wmma_bf16(pa, bv, O[dt]);
    }
  }

  // Normalize and write bf16 back in [B,T,C] layout for the output projection
#pragma unroll
  for (int r = 0; r < 8; r++) {
    int t = q_base + 8 * hlf + r;
    float inv_l = 1.0f / l_i[r];
    size_t base = ((size_t)(b * NT + t)) * NC + h * ND;
#pragma unroll
    for (int dt = 0; dt < 4; dt++)
      Y[base + dt * 16 + l15] = f2bf(O[dt][r] * inv_l);
  }
}

// ---------------------------------------------------------------------------
// Kernel 3: out = Y @ w_proj + b_proj  (Y is bf16 -> async copy to LDS;
// weights f32 -> bf16 on the fly)
// ---------------------------------------------------------------------------
__global__ __launch_bounds__(256)
void proj_gemm_kernel(const __bf16* __restrict__ Y, const float* __restrict__ w,
                      const float* __restrict__ bias, float* __restrict__ out) {
  __shared__ __attribute__((aligned(16))) __bf16 As[128][40];
  __shared__ __attribute__((aligned(16))) __bf16 Bs[128][40];
  const int tid = threadIdx.x;
  const int lane = tid & 31, wv = tid >> 5;
  const int hlf = lane >> 4, l15 = lane & 15;
  const int wm = wv >> 1, wn = wv & 1;
  const int row0 = blockIdx.x * 128;
  const int n0   = blockIdx.y * 128;

  v8f acc[2][4];
#pragma unroll
  for (int i = 0; i < 2; i++)
#pragma unroll
    for (int j = 0; j < 4; j++) acc[i][j] = vzero8();

  for (int k0 = 0; k0 < NC; k0 += 32) {
    // Stage A tile 128x32 bf16: ASYNCcnt-tracked global->LDS DMA (no VGPRs)
#pragma unroll
    for (int i = 0; i < 2; i++) {
      int idx = tid + i * 256;            // 0..511 chunks of 8 bf16 (16B)
      int r = idx >> 2, c = (idx & 3) * 8;
      const __bf16* gp = &Y[(size_t)(row0 + r) * NC + k0 + c];
      __bf16* lp = &As[r][c];
#if HAVE_ASYNC_LDS
      __builtin_amdgcn_global_load_async_to_lds_b128(
          (gptr_v4i)(void*)gp, (lptr_v4i)(void*)lp, 0, 0);
#else
      *(v8bf*)lp = *(const v8bf*)gp;
#endif
    }
    // Stage B tile 32x128 transposed into Bs[n][k]; batch loads then cvt+store
    float4 fb[4];
#pragma unroll
    for (int i = 0; i < 4; i++) {
      int idx = tid + i * 256;
      fb[i] = *(const float4*)&w[(size_t)(k0 + (idx >> 5)) * NC + n0 +
                                 (idx & 31) * 4];
    }
#pragma unroll
    for (int i = 0; i < 4; i++) {
      int idx = tid + i * 256;
      int kk = idx >> 5, n4 = (idx & 31) * 4;
      Bs[n4 + 0][kk] = f2bf(fb[i].x);
      Bs[n4 + 1][kk] = f2bf(fb[i].y);
      Bs[n4 + 2][kk] = f2bf(fb[i].z);
      Bs[n4 + 3][kk] = f2bf(fb[i].w);
    }
#if HAVE_ASYNC_LDS
    wait_async0();
#endif
    __syncthreads();
    if (k0 + 32 < NC) {
      __builtin_prefetch(&Y[(size_t)(row0 + (tid >> 1)) * NC + k0 + 32], 0, 1);
      __builtin_prefetch(&w[(size_t)(k0 + 32 + lane) * NC + n0], 0, 1);
    }
    v16bf af[2], bfr[4];
#pragma unroll
    for (int mt = 0; mt < 2; mt++) {
      int r = wm * 32 + mt * 16 + l15;
      af[mt] = mk_frag(&As[r][16 * hlf], &As[r][16 * hlf + 8]);
    }
#pragma unroll
    for (int nt = 0; nt < 4; nt++) {
      int n = wn * 64 + nt * 16 + l15;
      bfr[nt] = mk_frag(&Bs[n][16 * hlf], &Bs[n][16 * hlf + 8]);
    }
#pragma unroll
    for (int mt = 0; mt < 2; mt++)
#pragma unroll
      for (int nt = 0; nt < 4; nt++)
        acc[mt][nt] = wmma_bf16(af[mt], bfr[nt], acc[mt][nt]);
    __syncthreads();
  }

#pragma unroll
  for (int mt = 0; mt < 2; mt++) {
#pragma unroll
    for (int nt = 0; nt < 4; nt++) {
      int ng = n0 + wn * 64 + nt * 16 + l15;
      float bv = bias[ng];
#pragma unroll
      for (int r = 0; r < 8; r++) {
        int mg = row0 + wm * 32 + mt * 16 + 8 * hlf + r;
        out[(size_t)mg * NC + ng] = acc[mt][nt][r] + bv;
      }
    }
  }
}

// ---------------------------------------------------------------------------
extern "C" void kernel_launch(void* const* d_in, const int* in_sizes, int n_in,
                              void* d_out, int out_size, void* d_ws, size_t ws_size,
                              hipStream_t stream) {
  const float* x      = (const float*)d_in[0];
  // d_in[1] = attn_mask (bool causal) -- handled analytically, unused
  const float* w_attn = (const float*)d_in[2];
  const float* b_attn = (const float*)d_in[3];
  const float* w_proj = (const float*)d_in[4];
  const float* b_proj = (const float*)d_in[5];
  float* out = (float*)d_out;

  const size_t headElems = (size_t)NB * NH * NT * ND;  // 8.39M elems
  __bf16* Q  = (__bf16*)d_ws;
  __bf16* K  = Q + headElems;
  __bf16* Vt = K + headElems;
  __bf16* Y  = Vt + headElems;  // total 64 MB of bf16 scratch

  qkv_gemm_kernel<<<dim3((NB * NT) / 128, (3 * NC) / 128), 256, 0, stream>>>(
      x, w_attn, b_attn, Q, K, Vt);
  attention_kernel<<<dim3(NT / 128, NB * NH), 256, 0, stream>>>(Q, K, Vt, Y);
  proj_gemm_kernel<<<dim3((NB * NT) / 128, NC / 128), 256, 0, stream>>>(
      Y, w_proj, b_proj, out);
}
